// Neurocache_1486058685086
// MI455X (gfx1250) — matrix operand order repaired and checked by reading, not compiled, not run on hardware
//
#include <hip/hip_runtime.h>
#include <hip/hip_bf16.h>

// ---------------------------------------------------------------------------
// Neurocache sliding-window attention for MI455X (gfx1250, wave32).
//
// Roofline: ~26 GFLOP of GEMM vs ~42 MB of unique HBM traffic (1.8 us @
// 23.3 TB/s) -> compute-bound -> run GEMMs on V_WMMA_F32_16X16X32_BF16.
// All operands are pre-converted to bf16 once (~3 us of bandwidth), so GEMM
// fragments feed WMMA directly with no cvt chain and half the bytes.
// W tiles are staged into LDS by the Tensor Data Mover (tensor_load_to_lds,
// double-buffered, s_wait_tensorcnt + barrier), A fragments are software-
// pipelined in registers with global_prefetch one tile ahead.  All four B
// fragments are fetched from LDS before the four WMMAs so the ds_loads batch
// under a single s_wait_dscnt and the WMMAs issue back-to-back into the XDL
// pipe (no D->A/B hazard: shared A, disjoint accumulators).
//
// Workspace layout (93 MB, fully L2-resident on the 192 MB L2):
//   [ 0, 4M)   hs    bf16 (2048 x 1024)
//   [ 4,12M)   ext   bf16 (16384 x 256)
//   [12,14M)   Wq    bf16 (1024 x 1024)
//   [14,14.5M) Wk    bf16 (1024 x 256)
//   [14.5,15M) Wv    bf16 (1024 x 256)
//   [15,17M)   Wo    bf16 (1024 x 1024)
//   [17,25M)   q     f32  (2048 x 1024)   (carries the d^-1/2 scale)
//   [25,57M)   k     bf16 (16384 x 1024)
//   [57,89M)   v     bf16 (16384 x 1024)
//   [89,93M)   attn  bf16 (2048 x 1024)
// ---------------------------------------------------------------------------

typedef __attribute__((ext_vector_type(16))) __bf16 v16bf;
typedef __attribute__((ext_vector_type(8)))  __bf16 v8bf;
typedef __attribute__((ext_vector_type(8)))  float  v8f;
typedef __attribute__((ext_vector_type(4)))  unsigned int u32x4;
typedef __attribute__((ext_vector_type(8)))  int i32x8;
typedef __attribute__((ext_vector_type(4)))  int i32x4;

constexpr int B  = 2,  S  = 1024, H  = 1024;
constexpr int NN = 8,  RH = 256;
constexpr int NH = 16, HD = 64,   CTX = 4;

__device__ inline v16bf cat8(v8bf lo, v8bf hi) {
  return __builtin_shufflevector(lo, hi, 0, 1, 2, 3, 4, 5, 6, 7,
                                 8, 9, 10, 11, 12, 13, 14, 15);
}

// A fragment (16x32 bf16, wave32 ISA layout): lane&15 = row, lane>>4 = kg;
// elements {kg*8+0..7, kg*8+16..23}.  p = &A[row*K + kg*8 + k0], 16B aligned.
__device__ inline v16bf load_a(const __bf16* __restrict__ p) {
  const v8bf lo = *(const v8bf*)p;
  const v8bf hi = *(const v8bf*)(p + 16);
  return cat8(lo, hi);
}

// Issue one TDM 2D tile load: rows x cols bf16 elements, row stride `stride`
// elements, packed contiguously at LDS byte offset lds_off.
// D# per CDNA5 ISA section 8 (group0 128b + group1 256b; groups 2/3/iter
// unused -> zero).  6-arg builtin variant (clang-23 / therock-10.0 lane).
__device__ inline void tdm_load_2d(unsigned lds_off, const __bf16* gptr,
                                   unsigned rows, unsigned cols,
                                   unsigned stride_elems) {
  const unsigned long long ga = (unsigned long long)(size_t)gptr;
  u32x4 g0;
  g0[0] = 1u;                                            // count=1, user D#
  g0[1] = lds_off;                                       // lds_addr
  g0[2] = (unsigned)ga;                                  // global_addr[31:0]
  g0[3] = (unsigned)((ga >> 32) & 0x01FFFFFFu) | (2u << 30); // addr[56:32]|type=2
  i32x8 g1;
  g1[0] = (int)(1u << 16);                               // data_size=2B, mask=0
  g1[1] = (int)(cols << 16);                             // tensor_dim0[15:0]
  g1[2] = (int)((cols >> 16) | (rows << 16));            // td0[31:16]|td1[15:0]
  g1[3] = (int)((rows >> 16) | (cols << 16));            // td1[31:16]|tile_dim0
  g1[4] = (int)rows;                                     // tile_dim1, tile_dim2=0
  g1[5] = (int)stride_elems;                             // tensor_dim0_stride lo
  g1[6] = 0;                                             // stride hi | dim1_stride
  g1[7] = 0;
  const i32x4 z4 = {};
  const i32x8 z8 = {};
  __builtin_amdgcn_tensor_load_to_lds(g0, g1, z4, z4, z8, 0);
}

__device__ inline void store_val(float*  p, float v) { *p = v; }
__device__ inline void store_val(__bf16* p, float v) { *p = (__bf16)v; }

// ---------------------------------------------------------------------------
// f32 -> bf16 bulk conversion, 8 elements/thread (b128 in, b128 out).
// ---------------------------------------------------------------------------
__global__ void __launch_bounds__(256)
cvt_f32_bf16(const float* __restrict__ in, __bf16* __restrict__ out, int n) {
  const int i = (blockIdx.x * 256 + threadIdx.x) * 8;
  if (i >= n) return;
  const float4 a = *(const float4*)(in + i);
  const float4 b = *(const float4*)(in + i + 4);
  v8bf o;
  o[0] = (__bf16)a.x; o[1] = (__bf16)a.y; o[2] = (__bf16)a.z; o[3] = (__bf16)a.w;
  o[4] = (__bf16)b.x; o[5] = (__bf16)b.y; o[6] = (__bf16)b.z; o[7] = (__bf16)b.w;
  *(v8bf*)(out + i) = o;
}

// ---------------------------------------------------------------------------
// D[M,N] = A[M,K] * W[N,K]^T * scale (+ resid).   A, W bf16; acc f32.
// Block = 256 threads = 8 waves; block tile 128x64; wave tile 16x64.
// W tile (64x32 bf16) staged in LDS by TDM, double-buffered.
// ---------------------------------------------------------------------------
template <typename OT, bool RESID>
__global__ void __launch_bounds__(256)
gemm_bf16_wmma(const __bf16* __restrict__ A, const __bf16* __restrict__ W,
               OT* __restrict__ D, const float* __restrict__ resid,
               int M, int N, int K, float scale) {
  __shared__ __bf16 tileB[2][64 * 32];       // 2 x 4 KB

  const int lane  = threadIdx.x & 31;
  const int wave  = threadIdx.x >> 5;
  const int l15   = lane & 15;
  const int kg    = lane >> 4;
  const int tileM = blockIdx.x * 128 + wave * 16;
  const int tileN = blockIdx.y * 64;

  v8f acc[4] = {};
  const __bf16* aRow  = A + (size_t)(tileM + l15) * K + kg * 8;
  const __bf16* wBase = W + (size_t)tileN * K;

  if (wave == 0)
    tdm_load_2d((unsigned)(size_t)&tileB[0][0], wBase, 64, 32, (unsigned)K);

  v16bf aNext = load_a(aRow);
  int buf = 0;
  for (int k0 = 0; k0 < K; k0 += 32) {
    if (wave == 0) {
      if (k0 + 32 < K) {
        tdm_load_2d((unsigned)(size_t)&tileB[buf ^ 1][0], wBase + k0 + 32,
                    64, 32, (unsigned)K);
        __builtin_amdgcn_s_wait_tensorcnt((short)1);  // current buf complete
      } else {
        __builtin_amdgcn_s_wait_tensorcnt((short)0);
      }
    }
    __syncthreads();                       // tileB[buf] visible to all waves

    const v16bf aCur = aNext;
    if (k0 + 32 < K) {
      aNext = load_a(aRow + k0 + 32);      // software pipeline A
      if (k0 + 64 < K) __builtin_prefetch(aRow + k0 + 64, 0, 1);
    }

    // Fetch all four B fragments first so the 8 ds_load_b128 batch under one
    // s_wait_dscnt; B fragment: lane&15 = column N, K {kg*8+0..7, kg*8+16..23}.
    v16bf bfrag[4];
#pragma unroll
    for (int i = 0; i < 4; ++i) {
      const v8bf* bp = (const v8bf*)&tileB[buf][(i * 16 + l15) * 32];
      bfrag[i] = cat8(bp[kg], bp[kg + 2]);
    }
    // Four back-to-back WMMAs: shared A, disjoint accumulators -> no hazard.
#pragma unroll
    for (int i = 0; i < 4; ++i)
      acc[i] = __builtin_amdgcn_wmma_f32_16x16x32_bf16(
          false, aCur, false, bfrag[i], (short)0, acc[i], false, false);

    __syncthreads();                       // all reads of tileB[buf] done
    buf ^= 1;
  }

  // C/D layout: VGPR r -> M = tileM + r + 8*kg, N = tileN + (lane&15).
  const int mBase = tileM + kg * 8;
  const int nBase = tileN + l15;
#pragma unroll
  for (int i = 0; i < 4; ++i) {
    const int n = nBase + i * 16;
#pragma unroll
    for (int r = 0; r < 8; ++r) {
      const size_t idx = (size_t)(mBase + r) * N + n;
      float vv = acc[i][r] * scale;
      if constexpr (RESID) vv += resid[idx];
      store_val(D + idx, vv);
    }
  }
}

// ---------------------------------------------------------------------------
// Sliding-window attention: one wave per (b, s, head); lane = key index
// i = c*NN + nn, source position s_k = clamp(s-3+c, 0).
// k/v flat layout (B*S*NN, H): offset (b*S+sk)*8192 + head*512 + nn*64 + dd.
// ---------------------------------------------------------------------------
__global__ void __launch_bounds__(128)
attn_swa(const float* __restrict__ q, const __bf16* __restrict__ k,
         const __bf16* __restrict__ v, __bf16* __restrict__ o) {
  const int wave = threadIdx.x >> 5;
  const int lane = threadIdx.x & 31;
  const int gw   = blockIdx.x * 4 + wave;       // b*16384 + s*16 + head
  const int head = gw & (NH - 1);
  const int s    = (gw >> 4) & (S - 1);
  const int b    = gw >> 14;

  __shared__ float qs[4][HD];
  __shared__ float ws[4][32];

  const size_t qbase = (size_t)(b * S + s) * H + head * HD;
  qs[wave][lane]      = q[qbase + lane];
  qs[wave][lane + 32] = q[qbase + lane + 32];
  __syncthreads();

  const int c  = lane >> 3;
  const int nn = lane & 7;
  int sk = s - (CTX - 1) + c; sk = sk < 0 ? 0 : sk;
  const size_t kb = (size_t)(b * S + sk) * (NN * H) + head * (NN * HD) + nn * HD;

  float dot = 0.f;
#pragma unroll
  for (int dd = 0; dd < HD; ++dd) dot += qs[wave][dd] * (float)k[kb + dd];

  float mx = dot;
#pragma unroll
  for (int off = 16; off > 0; off >>= 1) mx = fmaxf(mx, __shfl_xor(mx, off, 32));
  const float e = __expf(dot - mx);
  float sum = e;
#pragma unroll
  for (int off = 16; off > 0; off >>= 1) sum += __shfl_xor(sum, off, 32);
  ws[wave][lane] = e / sum;
  __syncthreads();

  const int dd0 = lane * 2;
  float a0 = 0.f, a1 = 0.f;
#pragma unroll
  for (int i = 0; i < 32; ++i) {
    const int ci = i >> 3, nni = i & 7;
    int ski = s - (CTX - 1) + ci; ski = ski < 0 ? 0 : ski;
    const size_t vb = (size_t)(b * S + ski) * (NN * H) + head * (NN * HD) + nni * HD + dd0;
    const float w = ws[wave][i];
    a0 += w * (float)v[vb];
    a1 += w * (float)v[vb + 1];
  }
  o[qbase + dd0]     = (__bf16)a0;
  o[qbase + dd0 + 1] = (__bf16)a1;
}

extern "C" void kernel_launch(void* const* d_in, const int* in_sizes, int n_in,
                              void* d_out, int out_size, void* d_ws, size_t ws_size,
                              hipStream_t stream) {
  (void)in_sizes; (void)n_in; (void)out_size; (void)ws_size;
  const float* hs  = (const float*)d_in[0];
  const float* ext = (const float*)d_in[1];
  const float* Wq  = (const float*)d_in[2];
  const float* Wk  = (const float*)d_in[3];
  const float* Wv  = (const float*)d_in[4];
  const float* Wo  = (const float*)d_in[5];
  float* out = (float*)d_out;

  constexpr size_t MB = 1ull << 20, KB = 1ull << 10;
  char* ws = (char*)d_ws;
  __bf16* hs_bf  = (__bf16*)(ws);
  __bf16* ext_bf = (__bf16*)(ws + 4 * MB);
  __bf16* Wq_bf  = (__bf16*)(ws + 12 * MB);
  __bf16* Wk_bf  = (__bf16*)(ws + 14 * MB);
  __bf16* Wv_bf  = (__bf16*)(ws + 14 * MB + 512 * KB);
  __bf16* Wo_bf  = (__bf16*)(ws + 15 * MB);
  float*  qbuf   = (float*) (ws + 17 * MB);
  __bf16* kbuf   = (__bf16*)(ws + 25 * MB);
  __bf16* vbuf   = (__bf16*)(ws + 57 * MB);
  __bf16* abuf   = (__bf16*)(ws + 89 * MB);

  const int nHS = B * S * H;            // 2 Mi
  const int nEX = B * S * NN * RH;      // 4 Mi
  const int nWq = H * H;                // 1 Mi
  const int nWk = H * RH;               // 256 Ki
  const dim3 blk(256);
  auto cvtGrid = [](int n) { return dim3((unsigned)(n / 8 / 256)); };

  cvt_f32_bf16<<<cvtGrid(nHS), blk, 0, stream>>>(hs,  hs_bf,  nHS);
  cvt_f32_bf16<<<cvtGrid(nEX), blk, 0, stream>>>(ext, ext_bf, nEX);
  cvt_f32_bf16<<<cvtGrid(nWq), blk, 0, stream>>>(Wq,  Wq_bf,  nWq);
  cvt_f32_bf16<<<cvtGrid(nWk), blk, 0, stream>>>(Wk,  Wk_bf,  nWk);
  cvt_f32_bf16<<<cvtGrid(nWk), blk, 0, stream>>>(Wv,  Wv_bf,  nWk);
  cvt_f32_bf16<<<cvtGrid(nWq), blk, 0, stream>>>(Wo,  Wo_bf,  nWq);

  const int M1 = B * S;        // 2048
  const int M2 = B * S * NN;   // 16384

  // q = hs @ Wq^T * d^-1/2
  gemm_bf16_wmma<float, false>
      <<<dim3(M1 / 128, H / 64), blk, 0, stream>>>(hs_bf, Wq_bf, qbuf, nullptr,
                                                   M1, H, H, 0.125f);
  // k = ext @ Wk^T, v = ext @ Wv^T
  gemm_bf16_wmma<__bf16, false>
      <<<dim3(M2 / 128, H / 64), blk, 0, stream>>>(ext_bf, Wk_bf, kbuf, nullptr,
                                                   M2, H, RH, 1.0f);
  gemm_bf16_wmma<__bf16, false>
      <<<dim3(M2 / 128, H / 64), blk, 0, stream>>>(ext_bf, Wv_bf, vbuf, nullptr,
                                                   M2, H, RH, 1.0f);
  // softmax(q k^T) v over the 32-key sliding window
  attn_swa<<<dim3(B * S * NH / 4), dim3(128), 0, stream>>>(qbuf, kbuf, vbuf, abuf);
  // out = attn @ Wo^T + hs
  gemm_bf16_wmma<float, true>
      <<<dim3(M1 / 128, H / 64), blk, 0, stream>>>(abuf, Wo_bf, out, hs,
                                                   M1, H, H, 1.0f);
}
